// BlendedModel_89678917140638
// MI455X (gfx1250) — matrix operand order repaired and checked by reading, not compiled, not run on hardware
//
#include <hip/hip_runtime.h>

typedef __attribute__((ext_vector_type(16))) _Float16 v16h;
typedef __attribute__((ext_vector_type(8)))  float    v8f;

#if __has_builtin(__builtin_amdgcn_global_load_async_to_lds_b32)
#define HAS_ASYNC_LDS 1
typedef int __attribute__((address_space(1))) gb32;   // global b32 word
typedef int __attribute__((address_space(3))) sb32;   // LDS b32 word
#else
#define HAS_ASYNC_LDS 0
#endif

static constexpr int kBatch   = 4;
static constexpr int kH       = 256;
static constexpr int kW       = 256;
static constexpr int kTensor  = kBatch * 3 * kH * kW;   // 786432 elems per output tensor
static constexpr int kRowStr  = 144;   // padded row: covers wid*16 + K (max 143), keeps 16B runs
static constexpr int kRows    = 16;    // dy rows: 0..8 real, 9..15 zero (match A-operand rows)
static constexpr int kDataW   = 136;   // 128 cols + 2*4 halo of real data per row
static constexpr int kTileN   = 3 * kRows * kRowStr;    // 6912 floats = 27648 B LDS

__device__ __forceinline__ float fastrcp(float v) { return __builtin_amdgcn_rcpf(v); }
__device__ __forceinline__ float sigmoidf(float v) {
    return fastrcp(1.0f + __expf(-v));
}

__global__ __launch_bounds__(256) void fog_blur_wmma(
    const float* __restrict__ x, const float* __restrict__ dep, const float* __restrict__ noi,
    const float* __restrict__ sigk,
    const float* __restrict__ a0, const float* __restrict__ b0,   // ch0 (blue params)
    const float* __restrict__ a1, const float* __restrict__ b1,   // ch1 (green params)
    const float* __restrict__ a2, const float* __restrict__ b2,   // ch2 (red params)
    float* __restrict__ out)
{
    __shared__ float tile[kTileN];

    const int tid  = threadIdx.x;
    const int lane = tid & 31;
    const int hlf  = lane >> 4;          // 16-lane half of the wave
    const int m    = lane & 15;          // this lane's center (B/D column) AND its dy row n
    const int wid  = tid >> 5;

    const int xh      = blockIdx.x & 1;              // which half-row of the image
    const int y       = (blockIdx.x >> 1) & (kH - 1);
    const int bb      = blockIdx.x >> 9;
    const int x0start = xh * 128;
    const int x0      = x0start + wid * 16;          // this wave's 16-pixel segment

    // ---- stage 16 rows x 3 channels x 144 cols: async-DMA real data, zero the padding ----
    for (int i = tid; i < kTileN; i += 256) {
        const int chunk = i / kRowStr;               // c*16 + n
        const int cc    = i - chunk * kRowStr;
        const int ch    = chunk >> 4;
        const int nn    = chunk & 15;
        const int grow  = y - 4 + nn;
        const int gcol  = x0start - 4 + cc;
        const bool inb  = (nn <= 8) && (grow >= 0) && (grow < kH) &&
                          (cc < kDataW) && (gcol >= 0) && (gcol < kW);
        if (inb) {
            const int gidx = ((bb * 3 + ch) * kH + grow) * kW + gcol;
#if HAS_ASYNC_LDS
            __builtin_amdgcn_global_load_async_to_lds_b32(
                (gb32*)&x[gidx], (sb32*)&tile[i], 0, 0);
#else
            tile[i] = x[gidx];
#endif
        } else {
            tile[i] = 0.0f;                          // zero-pad: boundary rows/cols
        }
    }
#if HAS_ASYNC_LDS
#if __has_builtin(__builtin_amdgcn_s_wait_asynccnt)
    __builtin_amdgcn_s_wait_asynccnt(0);
#else
    asm volatile("s_wait_asynccnt 0x0" ::: "memory");
#endif
#endif
    __syncthreads();

    // ---- per-center sigma and 1D gaussian taps (4 exps instead of 81) ----
    const int   xm   = x0 + m;
    const float depc = dep[(bb * kH + y) * kW + xm];
    float sk = *sigk + 0.001f; sk = sk > 0.f ? sk : 0.f;
    const float sig = sk * depc;
    const float inv = -0.5f * fastrcp(sig * sig);
    const float g1 = __expf(inv * 1.f),  g2 = __expf(inv * 4.f);
    const float g3 = __expf(inv * 9.f),  g4 = __expf(inv * 16.f);
    const float S  = 1.f + 2.f * (g1 + g2 + g3 + g4);

    // ---- B operand (32x16 f16 weights, shared by channels): B[K,m] = g_m(|K-4-m|) ----
    // B layout: lane owns column N = lane&15; elems i<8 <-> K = i + 8*hlf, i>=8 <-> K = 16 + (i-8) + 8*hlf.
    // Zero for K outside [m, m+8] -- in particular ALL of K=24..31.
    v16h Bw;
#pragma unroll
    for (int i = 0; i < 16; ++i) {
        int K   = i + 8 * hlf + (i >= 8 ? 8 : 0);
        int rel = K - m;
        int d   = rel - 4; d = d < 0 ? -d : d;
        float w = (rel >= 0 && rel <= 8)
                    ? (d == 0 ? 1.f : d == 1 ? g1 : d == 2 ? g2 : d == 3 ? g3 : g4)
                    : 0.f;
        Bw[i] = (_Float16)w;
    }

    const float aC[3] = { sigmoidf(*a0), sigmoidf(*a1), sigmoidf(*a2) };
    const float bC[3] = { sigmoidf(*b0), sigmoidf(*b1), sigmoidf(*b2) };

    const int   n     = m;                           // dy row this lane packs into A
    const float noic  = noi[(bb * kH + y) * kW + xm];
    const float invS2 = fastrcp(S * S);
    const float wlo[8] = { g4, g3, g2, g1, 1.f, g1, g2, g3 };   // gv(|n-4|), n=0..7

#pragma unroll
    for (int c = 0; c < 3; ++c) {
        // ---- A operand (16x32 f16 data): A[n,K] = tile row n, col wid*16+K ----
        // Unconditional reads: boundary padding is zeroed in LDS; K>=24 columns may hold
        // stale neighbor data but B is identically zero there; rows 9..15 are zeroed and
        // their D rows are never consumed.
        // A layout: lanes 0-15 hold K=0..7,16..23; lanes 16-31 hold K=8..15,24..31.
        const int base = (c * kRows + n) * kRowStr + wid * 16 + hlf * 8;
        v16h Am;
#pragma unroll
        for (int i = 0; i < 8; ++i) {                // K = i + 8*hlf     (16B-aligned run)
            Am[i] = (_Float16)tile[base + i];
        }
#pragma unroll
        for (int i = 0; i < 8; ++i) {                // K = 16 + i + 8*hlf (16B-aligned run)
            Am[8 + i] = (_Float16)tile[base + 16 + i];
        }

        v8f Cz = {0.f, 0.f, 0.f, 0.f, 0.f, 0.f, 0.f, 0.f};
        // D'[n, m] = sum_dx g_m(|dx|) * x_c[y-4+n, x0+m+dx]
        v8f D = __builtin_amdgcn_wmma_f32_16x16x32_f16(
                    false, Am, false, Bw, (short)0, Cz, false, false);

        // ---- vertical pass: local FMAs with own taps; D rows M = r + 8*hlf = dy ----
        float p;
        if (hlf == 0) {
            p = 0.f;
#pragma unroll
            for (int r = 0; r < 8; ++r) p += wlo[r] * D[r];      // dy = 0..7
        } else {
            p = g4 * D[0];                                       // dy = 8 (only valid row)
        }
        p += __shfl_xor(p, 16, 32);                              // join the two halves

        const float blur  = p * invS2;
        const float t     = __expf(-aC[c] * depc);
        const float omt   = 1.f - t;
        const float bcst  = bC[c];
        const int   idx   = ((bb * 3 + c) * kH + y) * kW + xm;
        const float back  = (bcst * (1.f - noic) + noic) * omt;
        const float cback = bcst * omt;
        if (lane < 16) {
            const float xc = x[idx];
            out[idx]           = blur * t + back;    // out
            out[kTensor + idx] = xc   * t + cback;   // clear_out
        } else {
            out[2 * kTensor + idx] = blur * t + cback;  // clear_forward
        }
    }
}

extern "C" void kernel_launch(void* const* d_in, const int* in_sizes, int n_in,
                              void* d_out, int out_size, void* d_ws, size_t ws_size,
                              hipStream_t stream) {
    (void)in_sizes; (void)n_in; (void)out_size; (void)d_ws; (void)ws_size;
    const float* x    = (const float*)d_in[0];
    const float* dep  = (const float*)d_in[1];
    const float* noi  = (const float*)d_in[2];
    const float* sigk = (const float*)d_in[3];
    // output ch0 uses (alpha_b,b_b)=d_in[8],[9]; ch1 (alpha_g,b_g)=d_in[6],[7];
    // ch2 (alpha_r,b_r)=d_in[4],[5]
    const float* a0 = (const float*)d_in[8];
    const float* b0 = (const float*)d_in[9];
    const float* a1 = (const float*)d_in[6];
    const float* b1 = (const float*)d_in[7];
    const float* a2 = (const float*)d_in[4];
    const float* b2 = (const float*)d_in[5];
    float* out = (float*)d_out;

    // block = one half-row (8 segments, 8 waves); grid = 4 batches * 256 rows * 2 halves
    dim3 grid(2048), block(256);
    fog_blur_wmma<<<grid, block, 0, stream>>>(x, dep, noi, sigk, a0, b0, a1, b1, a2, b2, out);
}
